// QuantTokenPredictor_24635932410273
// MI455X (gfx1250) — matrix operand order repaired and checked by reading, not compiled, not run on hardware
//
#include <hip/hip_runtime.h>
#include <hip/hip_bf16.h>

typedef __attribute__((ext_vector_type(16))) _Float16 v16h;
typedef __attribute__((ext_vector_type(8)))  _Float16 v8h;
typedef __attribute__((ext_vector_type(8)))  float    v8f;
typedef __attribute__((ext_vector_type(4)))  unsigned int v4u;   // 16-byte vector
typedef __attribute__((ext_vector_type(4)))  int v4i;

// Problem constants (reference: h (2,2048,512) f32, weight (32000,512) f32)
#define MDIM 4096
#define NDIM 32000
#define KDIM 512
#define KC   32      // K chunk consumed per WMMA step
#define BM   128     // block tile M
#define BN   128     // block tile N
#define PAD  40      // LDS row stride in halves (80B = 16B-aligned, bank-conflict free)
#define NCHUNK (KDIM / KC)   // 16

// gfx1250 async global->LDS path (ASYNCcnt-tracked), if the toolchain exposes it.
#if __has_builtin(__builtin_amdgcn_global_load_async_to_lds_b128) && \
    __has_builtin(__builtin_amdgcn_s_wait_asynccnt)
#define QTP_ASYNC 1
#else
#define QTP_ASYNC 0
#endif

// Builtin signature (from compiler diagnostic): (v4i AS1* src, v4i AS3* dst, imm, imm)
#define QTP_AS1(p) ((__attribute__((address_space(1))) v4i*)(p))
#define QTP_AS3(p) ((__attribute__((address_space(3))) v4i*)(p))

// ---------------------------------------------------------------------------
// Pass 1: L2-normalize h rows -> f16, and per-token ||hn||^2
// ---------------------------------------------------------------------------
__global__ void qtp_norm_h(const float* __restrict__ h,
                           _Float16* __restrict__ hf,
                           float* __restrict__ hsq) {
    __shared__ float ssum;
    const int tok = blockIdx.x;            // one token per block
    const int tid = threadIdx.x;           // 512 threads, one element each
    float x = h[(size_t)tok * KDIM + tid];
    float s = x * x;
    #pragma unroll
    for (int m = 16; m >= 1; m >>= 1) s += __shfl_xor(s, m, 32);
    if (tid == 0) ssum = 0.0f;
    __syncthreads();
    if ((tid & 31) == 0) atomicAdd(&ssum, s);
    __syncthreads();
    const float tot = ssum;
    const float inv = 1.0f / fmaxf(sqrtf(tot), 1e-12f);
    hf[(size_t)tok * KDIM + tid] = (_Float16)(x * inv);
    if (tid == 0) hsq[tok] = tot * inv * inv;   // == ||hn||^2 (~1.0)
}

// ---------------------------------------------------------------------------
// Pass 2: convert codebook rows -> f16, and per-row ||w||^2
// ---------------------------------------------------------------------------
__global__ void qtp_conv_w(const float* __restrict__ w,
                           _Float16* __restrict__ wf,
                           float* __restrict__ wsq) {
    __shared__ float ssum;
    const int row = blockIdx.x;            // one codebook row per block
    const int tid = threadIdx.x;           // 128 threads, 4 elements each
    float s = 0.0f;
    #pragma unroll
    for (int i = 0; i < 4; ++i) {
        const int k = tid + i * 128;
        const float x = w[(size_t)row * KDIM + k];
        s += x * x;
        wf[(size_t)row * KDIM + k] = (_Float16)x;
    }
    #pragma unroll
    for (int m = 16; m >= 1; m >>= 1) s += __shfl_xor(s, m, 32);
    if (tid == 0) ssum = 0.0f;
    __syncthreads();
    if ((tid & 31) == 0) atomicAdd(&ssum, s);
    __syncthreads();
    if (tid == 0) wsq[row] = ssum;
}

// Build a v16h WMMA fragment from two 16-byte LDS chunks (2x ds_load_b128).
static __device__ __forceinline__ v16h ld_frag(const _Float16* p0, const _Float16* p1) {
    v8h lo = *(const v8h*)p0;
    v8h hi = *(const v8h*)p1;
    v16h r;
    #pragma unroll
    for (int i = 0; i < 8; ++i) { r[i] = lo[i]; r[i + 8] = hi[i]; }
    return r;
}

// ---------------------------------------------------------------------------
// Pass 3: fused distance GEMM.
//   out[m,n] = sqrt(max(hsq[m] - 2*(hn . w_n) + wsq[n], 0))
// 256 threads = 8 wave32, block tile 128x128, wave tile 64x32 (4x2 WMMA accs).
// Double-buffered LDS over K in chunks of 32; async global->LDS when available.
// ---------------------------------------------------------------------------
__global__ __launch_bounds__(256)
void qtp_dist_gemm(const _Float16* __restrict__ A,   // (MDIM, KDIM) normalized h, f16
                   const _Float16* __restrict__ B,   // (NDIM, KDIM) codebook, f16
                   const float* __restrict__ hsq,
                   const float* __restrict__ wsq,
                   float* __restrict__ out) {
    __shared__ _Float16 As[2][BM * PAD];
    __shared__ _Float16 Bs[2][BN * PAD];

    const int tid   = threadIdx.x;
    const int lane  = tid & 31;
    const int wave  = tid >> 5;
    const int waveM = wave & 1;        // 2 waves along M (64 rows each)
    const int waveN = wave >> 1;       // 4 waves along N (32 cols each)
    const int half  = lane >> 4;       // lane group (0: lanes 0-15, 1: lanes 16-31)
    const int r     = lane & 15;

    const int mBase = blockIdx.y * BM;
    const int nBase = blockIdx.x * BN;

    // Loader: tile chunk is 128 rows x 32 halves; split into 512 16B segments,
    // 2 per thread per matrix. segment s -> row s>>2, 8-half column (s&3)*8.
    const int s0  = tid, s1 = tid + 256;
    const int ar0 = s0 >> 2, ac0 = (s0 & 3) * 8;
    const int ar1 = s1 >> 2, ac1 = (s1 & 3) * 8;

    const _Float16* aTile = A + (size_t)mBase * KDIM;
    const _Float16* bTile = B + (size_t)nBase * KDIM;

#if QTP_ASYNC
    // Direct global->LDS DMA per lane, tracked by ASYNCcnt: no VGPR staging.
    auto gasync = [&](int kc, int buf) {
        const _Float16* ap = aTile + kc * KC;
        const _Float16* bp = bTile + kc * KC;
        __builtin_amdgcn_global_load_async_to_lds_b128(
            QTP_AS1(ap + (size_t)ar0 * KDIM + ac0), QTP_AS3(&As[buf][ar0 * PAD + ac0]), 0, 0);
        __builtin_amdgcn_global_load_async_to_lds_b128(
            QTP_AS1(ap + (size_t)ar1 * KDIM + ac1), QTP_AS3(&As[buf][ar1 * PAD + ac1]), 0, 0);
        __builtin_amdgcn_global_load_async_to_lds_b128(
            QTP_AS1(bp + (size_t)ar0 * KDIM + ac0), QTP_AS3(&Bs[buf][ar0 * PAD + ac0]), 0, 0);
        __builtin_amdgcn_global_load_async_to_lds_b128(
            QTP_AS1(bp + (size_t)ar1 * KDIM + ac1), QTP_AS3(&Bs[buf][ar1 * PAD + ac1]), 0, 0);
    };
#else
    v4u ra0, ra1, rb0, rb1;
    auto gload = [&](int kc) {
        const _Float16* ap = aTile + kc * KC;
        const _Float16* bp = bTile + kc * KC;
        ra0 = *(const v4u*)(ap + (size_t)ar0 * KDIM + ac0);
        ra1 = *(const v4u*)(ap + (size_t)ar1 * KDIM + ac1);
        rb0 = *(const v4u*)(bp + (size_t)ar0 * KDIM + ac0);
        rb1 = *(const v4u*)(bp + (size_t)ar1 * KDIM + ac1);
    };
    auto lstore = [&](int buf) {
        *(v4u*)&As[buf][ar0 * PAD + ac0] = ra0;
        *(v4u*)&As[buf][ar1 * PAD + ac1] = ra1;
        *(v4u*)&Bs[buf][ar0 * PAD + ac0] = rb0;
        *(v4u*)&Bs[buf][ar1 * PAD + ac1] = rb1;
    };
#endif

    v8f acc[4][2];
    #pragma unroll
    for (int tm = 0; tm < 4; ++tm)
        #pragma unroll
        for (int tn = 0; tn < 2; ++tn)
            #pragma unroll
            for (int g = 0; g < 8; ++g) acc[tm][tn][g] = 0.0f;

#if QTP_ASYNC
    gasync(0, 0);
    __builtin_amdgcn_s_wait_asynccnt(0);
    __syncthreads();
#else
    gload(0);
    lstore(0);
    __syncthreads();
#endif

    for (int kc = 0; kc < NCHUNK; ++kc) {
        const int cur = kc & 1;
        if (kc + 1 < NCHUNK) {
#if QTP_ASYNC
            gasync(kc + 1, cur ^ 1);             // DMA in flight during WMMAs
#else
            gload(kc + 1);                       // loads in flight during WMMAs
#endif
            if (kc + 2 < NCHUNK) {               // -> global_prefetch_b8
                __builtin_prefetch(aTile + (kc + 2) * KC + (size_t)ar0 * KDIM, 0, 0);
                __builtin_prefetch(bTile + (kc + 2) * KC + (size_t)ar0 * KDIM, 0, 0);
            }
        }

        // A fragment (16x32 f16): lanes 0-15 hold K{0-7,16-23}, lanes 16-31 K{8-15,24-31}
        const _Float16* Ab = As[cur];
        const _Float16* Bb = Bs[cur];
        const int ka = half * 8;
        v16h af[4], bfr[2];
        #pragma unroll
        for (int tm = 0; tm < 4; ++tm) {
            const int row = waveM * 64 + tm * 16 + r;
            af[tm] = ld_frag(Ab + row * PAD + ka, Ab + row * PAD + ka + 16);
        }
        // B fragment (32x16 f16): lanes 0-15 hold K{0-7,16-23}, lanes 16-31 K{8-15,24-31},
        // column N = lane&15 -> codebook row (row-major in LDS, K contiguous)
        #pragma unroll
        for (int tn = 0; tn < 2; ++tn) {
            const int row = waveN * 32 + tn * 16 + r;
            const _Float16* p = Bb + row * PAD + half * 8;
            bfr[tn] = ld_frag(p, p + 16);
        }
        #pragma unroll
        for (int tm = 0; tm < 4; ++tm)
            #pragma unroll
            for (int tn = 0; tn < 2; ++tn)
                acc[tm][tn] = __builtin_amdgcn_wmma_f32_16x16x32_f16(
                    false, af[tm], false, bfr[tn],
                    (short)0, acc[tm][tn], false, false);

        if (kc + 1 < NCHUNK) {
#if QTP_ASYNC
            __builtin_amdgcn_s_wait_asynccnt(0);  // next buffer resident in LDS
#else
            lstore(cur ^ 1);
#endif
        }
        __syncthreads();
    }

    // Epilogue: C/D layout -> VGPR g: lanes0-15 (M=g,N=lane), lanes16-31 (M=g+8,N=lane-16)
    #pragma unroll
    for (int tm = 0; tm < 4; ++tm) {
        #pragma unroll
        for (int tn = 0; tn < 2; ++tn) {
            const int mrow = mBase + waveM * 64 + tm * 16 + half * 8;
            const int col  = nBase + waveN * 32 + tn * 16 + r;
            const float wq = wsq[col];
            #pragma unroll
            for (int g = 0; g < 8; ++g) {
                const int row = mrow + g;
                const float d = hsq[row] - 2.0f * acc[tm][tn][g] + wq;
                out[(size_t)row * NDIM + col] = sqrtf(fmaxf(d, 0.0f));
            }
        }
    }
}

// ---------------------------------------------------------------------------
extern "C" void kernel_launch(void* const* d_in, const int* in_sizes, int n_in,
                              void* d_out, int out_size, void* d_ws, size_t ws_size,
                              hipStream_t stream) {
    (void)in_sizes; (void)n_in; (void)out_size; (void)ws_size;
    const float* h = (const float*)d_in[0];   // (2,2048,512) f32
    const float* w = (const float*)d_in[1];   // (32000,512) f32
    float* out = (float*)d_out;               // (2,2048,32000) f32

    // Workspace layout (~37 MB):
    _Float16* hF  = (_Float16*)d_ws;                       // MDIM*KDIM f16
    _Float16* wF  = hF + (size_t)MDIM * KDIM;              // NDIM*KDIM f16
    float*    hsq = (float*)(wF + (size_t)NDIM * KDIM);    // MDIM f32
    float*    wsq = hsq + MDIM;                            // NDIM f32

    qtp_norm_h<<<MDIM, 512, 0, stream>>>(h, hF, hsq);
    qtp_conv_w<<<NDIM, 128, 0, stream>>>(w, wF, wsq);

    dim3 grid(NDIM / BN, MDIM / BM);   // 250 x 32
    qtp_dist_gemm<<<grid, 256, 0, stream>>>(hF, wF, hsq, wsq, out);
}
